// ImprovedGCN_67542655697039
// MI455X (gfx1250) — compile-verified, matrix operand order checked by reading, not run on hardware
//
#include <hip/hip_runtime.h>
#include <hip/hip_bf16.h>
#include <stdint.h>

#define EPS        1e-5f
#define NEG_SLOPE  0.2f

typedef __attribute__((ext_vector_type(16))) __bf16 v16bf;
typedef __attribute__((ext_vector_type(8)))  float  v8f;

static inline int    cdiv(int a, int b)            { return (a + b - 1) / b; }
static inline size_t alignUp(size_t x, size_t a)   { return (x + a - 1) / a * a; }

// ---------------------------------------------------------------------------
// GEMM: C[M,128] = A_bf16[M,128] @ W[128,128] (+bias), W given pre-transposed
// as BT[n][k] (row-major 128x128 bf16). One wave -> one 16x16 tile, K in 4
// steps of 32 via v_wmma_f32_16x16x32_bf16. 8 waves/block cover 128 cols.
// CDNA5 16-bit A fragment: lane L<16 holds row L, K {k..k+7, k+16..k+23};
// lane L>=16 holds row L-16, K {k+8..k+15, k+24..k+31}. B mirrors with
// lane = column index, sourced from W^T rows.
// ---------------------------------------------------------------------------
__global__ void k_gemm_bf16(const __bf16* __restrict__ A,
                            const __bf16* __restrict__ BT,
                            const float*  __restrict__ bias,
                            float*        __restrict__ C,
                            int M)
{
    const int wave = threadIdx.x >> 5;
    const int lane = threadIdx.x & 31;
    const int half = lane >> 4;
    const int r    = lane & 15;
    const int m0   = blockIdx.x * 16;
    const int n0   = wave * 16;
    if (m0 >= M) return;

    const __bf16* ap = A  + (size_t)(m0 + r) * 128 + half * 8;
    const __bf16* bp = BT + (size_t)(n0 + r) * 128 + half * 8;

    v8f c = {0.f, 0.f, 0.f, 0.f, 0.f, 0.f, 0.f, 0.f};
#pragma unroll
    for (int k = 0; k < 128; k += 32) {
        v16bf a, b;
        ((uint4*)&a)[0] = *(const uint4*)(ap + k);
        ((uint4*)&a)[1] = *(const uint4*)(ap + k + 16);
        ((uint4*)&b)[0] = *(const uint4*)(bp + k);
        ((uint4*)&b)[1] = *(const uint4*)(bp + k + 16);
        c = __builtin_amdgcn_wmma_f32_16x16x32_bf16(false, a, false, b,
                                                    (short)0, c, false, false);
    }
    // C/D layout: VGPR v, lane<16 -> (row v, col lane); lane>=16 -> (row v+8)
    const int   col = n0 + r;
    const float bv  = bias ? bias[col] : 0.f;
    float* out = C + (size_t)(m0 + half * 8) * 128 + col;
#pragma unroll
    for (int v = 0; v < 8; ++v)
        out[(size_t)v * 128] = c[v] + bv;
}

// --------------------- attention scores s = h.a_src, d = h.a_dst ------------
__global__ void k_scores(const float* __restrict__ h,
                         const float* __restrict__ asrc,
                         const float* __restrict__ adst,
                         float* __restrict__ s, float* __restrict__ d, int N)
{
    int wave = threadIdx.x >> 5, lane = threadIdx.x & 31;
    int n = blockIdx.x * 8 + wave;
    if (n >= N) return;
    float4 hv = *(const float4*)(h + (size_t)n * 128 + lane * 4);
    float4 as = *(const float4*)(asrc + lane * 4);
    float4 ad = *(const float4*)(adst + lane * 4);
    float ss = hv.x * as.x + hv.y * as.y + hv.z * as.z + hv.w * as.w;
    float dd = hv.x * ad.x + hv.y * ad.y + hv.z * ad.z + hv.w * ad.w;
    for (int off = 16; off; off >>= 1) {
        ss += __shfl_xor(ss, off, 32);
        dd += __shfl_xor(dd, off, 32);
    }
    if (lane == 0) { s[n] = ss; d[n] = dd; }
}

// --------------------- CSR build (dst is layer-invariant: build once) -------
__global__ void k_count_init(int* counts, int* cursor, int N)
{
    int i = blockIdx.x * blockDim.x + threadIdx.x;
    if (i < N) { counts[i] = 1; cursor[i] = 0; }   // 1 = self loop
}
__global__ void k_count_edges(const int* __restrict__ ei, int E, int* counts)
{
    int e = blockIdx.x * blockDim.x + threadIdx.x;
    if (e < E) atomicAdd(&counts[ei[E + e]], 1);
}
__global__ void k_scan1(const int* __restrict__ counts, int* __restrict__ rowptr,
                        int* __restrict__ partials, int N)
{
    __shared__ int sh[256];
    int t = threadIdx.x, g = blockIdx.x * 256 + t;
    int v = (g < N) ? counts[g] : 0;
    sh[t] = v;
    for (int off = 1; off < 256; off <<= 1) {
        __syncthreads();
        int tv = (t >= off) ? sh[t - off] : 0;
        __syncthreads();
        sh[t] += tv;
    }
    __syncthreads();
    if (g < N) rowptr[g] = sh[t] - v;             // exclusive
    if (t == 255) partials[blockIdx.x] = sh[255];
}
__global__ void k_scan2(int* partials, int nb, int* rowptr, int N)
{
    if (threadIdx.x == 0) {
        int acc = 0;
        for (int i = 0; i < nb; i++) { int tv = partials[i]; partials[i] = acc; acc += tv; }
        rowptr[N] = acc;
    }
}
__global__ void k_scan3(int* rowptr, const int* __restrict__ partials, int N)
{
    int g = blockIdx.x * 256 + threadIdx.x;
    if (g < N) rowptr[g] += partials[blockIdx.x];
}
__global__ void k_scatter(const int* __restrict__ ei, int E, int N,
                          const int* __restrict__ rowptr, int* __restrict__ cursor,
                          int* __restrict__ srcs)
{
    int i = blockIdx.x * blockDim.x + threadIdx.x;
    if (i >= E + N) return;
    int src, dst;
    if (i < E) { src = ei[i]; dst = ei[E + i]; } else { src = dst = i - E; }
    int pos = rowptr[dst] + atomicAdd(&cursor[dst], 1);
    srcs[pos] = src;
}

// --------------------- GAT softmax-aggregate: one wave per dst node ---------
__global__ void k_gat_aggregate(const int* __restrict__ rowptr,
                                const int* __restrict__ srcs,
                                const float* __restrict__ s,
                                const float* __restrict__ d,
                                const float* __restrict__ h,
                                const float* __restrict__ bias,
                                float* __restrict__ out, int N)
{
    int wave = threadIdx.x >> 5, lane = threadIdx.x & 31;
    int n = blockIdx.x * 8 + wave;
    if (n >= N) return;
    int beg = rowptr[n], end = rowptr[n + 1];
    float di = d[n];

    // pass 1: segment max (lanes over edges)
    float m = -3.4e38f;
    for (int j = beg + lane; j < end; j += 32) {
        int sj = srcs[j];
        __builtin_prefetch(h + (size_t)sj * 128, 0, 1);   // warm L2 for pass 3
        float e = s[sj] + di;
        e = (e > 0.f) ? e : NEG_SLOPE * e;
        m = fmaxf(m, e);
    }
    for (int off = 16; off; off >>= 1) m = fmaxf(m, __shfl_xor(m, off, 32));

    // pass 2: sum of exp
    float z = 0.f;
    for (int j = beg + lane; j < end; j += 32) {
        float e = s[srcs[j]] + di;
        e = (e > 0.f) ? e : NEG_SLOPE * e;
        z += __expf(e - m);
    }
    for (int off = 16; off; off >>= 1) z += __shfl_xor(z, off, 32);
    float invz = 1.f / z;

    // pass 3: weighted row gather (lanes over features, edges serial)
    int f = lane * 4;
    float4 acc = {0.f, 0.f, 0.f, 0.f};
    for (int j = beg; j < end; ++j) {
        int sj = srcs[j];
        float e = s[sj] + di;
        e = (e > 0.f) ? e : NEG_SLOPE * e;
        float alpha = __expf(e - m) * invz;
        float4 hv = *(const float4*)(h + (size_t)sj * 128 + f);
        acc.x += alpha * hv.x; acc.y += alpha * hv.y;
        acc.z += alpha * hv.z; acc.w += alpha * hv.w;
    }
    float4 bv = *(const float4*)(bias + f);
    acc.x += bv.x; acc.y += bv.y; acc.z += bv.z; acc.w += bv.w;
    *(float4*)(out + (size_t)n * 128 + f) = acc;
}

// --------------------- BatchNorm (training-mode batch stats) ----------------
__global__ void k_zero(float* p, int n)
{
    int i = blockIdx.x * blockDim.x + threadIdx.x;
    if (i < n) p[i] = 0.f;
}
__global__ void k_bn_stats(const float* __restrict__ x, float* __restrict__ stats, int N)
{
    __shared__ float ssum[128], ssq[128];
    int f = threadIdx.x & 127;
    int sub = threadIdx.x >> 7;
    int r0 = blockIdx.x * 256 + sub;
    int rend = min(blockIdx.x * 256 + 256, N);
    float sum = 0.f, sq = 0.f;
    for (int r = r0; r < rend; r += 2) {
        float v = x[(size_t)r * 128 + f];
        sum += v; sq += v * v;
    }
    if (sub == 0) { ssum[f] = sum; ssq[f] = sq; }
    __syncthreads();
    if (sub == 1) { ssum[f] += sum; ssq[f] += sq; }
    __syncthreads();
    if (sub == 0) {
        atomicAdd(&stats[f], ssum[f]);
        atomicAdd(&stats[128 + f], ssq[f]);
    }
}
__global__ void k_bn_finalize(const float* __restrict__ stats,
                              const float* __restrict__ gamma,
                              const float* __restrict__ beta,
                              float* __restrict__ scsh, float invN)
{
    int f = threadIdx.x;
    float mean = stats[f] * invN;
    float var  = stats[128 + f] * invN - mean * mean;   // biased, matches jnp.var
    float sc   = gamma[f] * rsqrtf(var + EPS);
    scsh[f]       = sc;
    scsh[128 + f] = beta[f] - mean * sc;
}
// apply BN (+ optional relu + residual in place) and emit bf16 for next GEMM
template <bool RELU_RES>
__global__ void k_bn_apply(const float* __restrict__ x, const float* __restrict__ scsh,
                           float* __restrict__ res, __bf16* __restrict__ hbf, int N)
{
    int i = blockIdx.x * blockDim.x + threadIdx.x;   // float4 index
    if (i >= N * 32) return;
    int f = (i & 31) * 4;
    float4 v  = *(const float4*)(x + (size_t)i * 4);
    float4 sc = *(const float4*)(scsh + f);
    float4 sh = *(const float4*)(scsh + 128 + f);
    v.x = sc.x * v.x + sh.x; v.y = sc.y * v.y + sh.y;
    v.z = sc.z * v.z + sh.z; v.w = sc.w * v.w + sh.w;
    if (RELU_RES) {
        float4 rv = *(const float4*)(res + (size_t)i * 4);
        v.x = fmaxf(v.x, 0.f) + rv.x; v.y = fmaxf(v.y, 0.f) + rv.y;
        v.z = fmaxf(v.z, 0.f) + rv.z; v.w = fmaxf(v.w, 0.f) + rv.w;
        *(float4*)(res + (size_t)i * 4) = v;          // residual = new h
    }
    union { __bf16 b[4]; uint2 u; } cv;
    cv.b[0] = (__bf16)v.x; cv.b[1] = (__bf16)v.y;
    cv.b[2] = (__bf16)v.z; cv.b[3] = (__bf16)v.w;
    *(uint2*)(hbf + (size_t)i * 4) = cv.u;
}

// --------------------- weight prep: f32 -> bf16, transposed -----------------
struct WtArgs { const float* w[8]; __bf16* wt[8]; };
__global__ void k_convert_weights(WtArgs a)
{
    int m = blockIdx.y, n = blockIdx.x, k = threadIdx.x;
    a.wt[m][n * 128 + k] = (__bf16)a.w[m][k * 128 + n];   // WT[n][k] = W[k][n]
}
__global__ void k_f32_to_bf16(const float* __restrict__ s, __bf16* __restrict__ d, int n)
{
    int i = blockIdx.x * blockDim.x + threadIdx.x;
    if (i < n) d[i] = (__bf16)s[i];
}

// ---------------------------------------------------------------------------
extern "C" void kernel_launch(void* const* d_in, const int* in_sizes, int n_in,
                              void* d_out, int out_size, void* d_ws, size_t ws_size,
                              hipStream_t stream)
{
    (void)n_in; (void)out_size; (void)ws_size;
    const int N  = in_sizes[0] / 128;   // 100000 (multiple of 16)
    const int E  = in_sizes[1] / 2;     // 1600000
    const int EN = E + N;

    const float* x      = (const float*)d_in[0];
    const int*   ei     = (const int*)d_in[1];
    const float* W0     = (const float*)d_in[2];
    const float* asrc0  = (const float*)d_in[3];
    const float* adst0  = (const float*)d_in[4];
    const float* b0     = (const float*)d_in[5];
    const float* Wm     = (const float*)d_in[6];
    const float* asrcm  = (const float*)d_in[7];
    const float* adstm  = (const float*)d_in[8];
    const float* bm     = (const float*)d_in[9];
    const float* W5     = (const float*)d_in[10];
    const float* asrc5  = (const float*)d_in[11];
    const float* adst5  = (const float*)d_in[12];
    const float* b5     = (const float*)d_in[13];
    const float* gammas = (const float*)d_in[14];
    const float* betas  = (const float*)d_in[15];
    const float* Wr     = (const float*)d_in[16];
    const float* br     = (const float*)d_in[17];
    const float* Wo     = (const float*)d_in[18];
    const float* bo     = (const float*)d_in[19];

    // ---- workspace carve (~190 MB) ----
    char* p = (char*)d_ws;
    auto take = [&](size_t bytes) -> char* { char* q = p; p += alignUp(bytes, 256); return q; };
    int*    srcs     = (int*)take((size_t)EN * 4);
    int*    rowptr   = (int*)take((size_t)(N + 1) * 4);
    int*    cursor   = (int*)take((size_t)N * 4);
    int*    counts   = (int*)take((size_t)N * 4);
    int     nb       = cdiv(N, 256);
    int*    partials = (int*)take((size_t)nb * 4);
    float*  sbuf     = (float*)take((size_t)N * 4);
    float*  dbuf     = (float*)take((size_t)N * 4);
    float*  stats    = (float*)take(256 * 4);
    float*  scsh     = (float*)take(256 * 4);
    __bf16* wtb      = (__bf16*)take((size_t)8 * 128 * 128 * 2);
    __bf16* hbf      = (__bf16*)take((size_t)N * 128 * 2);
    float*  resb     = (float*)take((size_t)N * 128 * 4);
    float*  hlin     = (float*)take((size_t)N * 128 * 4);
    float*  agg      = (float*)take((size_t)N * 128 * 4);

    __bf16* WrT = wtb;
    __bf16* WT[6] = { wtb + 16384, wtb + 2 * 16384, wtb + 3 * 16384,
                      wtb + 4 * 16384, wtb + 5 * 16384, wtb + 6 * 16384 };
    __bf16* WoT = wtb + 7 * 16384;

    WtArgs wa;
    wa.w[0] = Wr; wa.wt[0] = WrT;
    wa.w[1] = W0; wa.wt[1] = WT[0];
    for (int i = 0; i < 4; i++) { wa.w[2 + i] = Wm + (size_t)i * 16384; wa.wt[2 + i] = WT[1 + i]; }
    wa.w[6] = W5; wa.wt[6] = WT[5];
    wa.w[7] = Wo; wa.wt[7] = WoT;
    k_convert_weights<<<dim3(128, 8), 128, 0, stream>>>(wa);
    k_f32_to_bf16<<<cdiv(N * 128, 256), 256, 0, stream>>>(x, hbf, N * 128);

    // CSR by destination (once — graph identical for all 6 layers)
    k_count_init <<<cdiv(N, 256), 256, 0, stream>>>(counts, cursor, N);
    k_count_edges<<<cdiv(E, 256), 256, 0, stream>>>(ei, E, counts);
    k_scan1<<<nb, 256, 0, stream>>>(counts, rowptr, partials, N);
    k_scan2<<<1, 32, 0, stream>>>(partials, nb, rowptr, N);
    k_scan3<<<nb, 256, 0, stream>>>(rowptr, partials, N);
    k_scatter<<<cdiv(EN, 256), 256, 0, stream>>>(ei, E, N, rowptr, cursor, srcs);

    // residual = x @ Wr + br
    k_gemm_bf16<<<N / 16, 256, 0, stream>>>(hbf, WrT, br, resb, N);

    const float* lasrc[6] = { asrc0, asrcm, asrcm + 128, asrcm + 256, asrcm + 384, asrc5 };
    const float* ladst[6] = { adst0, adstm, adstm + 128, adstm + 256, adstm + 384, adst5 };
    const float* lb[6]    = { b0, bm, bm + 128, bm + 256, bm + 384, b5 };

    const float invN = 1.0f / (float)N;
    const int nodeBlocks = cdiv(N, 8);

    for (int L = 0; L < 6; ++L) {
        k_gemm_bf16<<<N / 16, 256, 0, stream>>>(hbf, WT[L], nullptr, hlin, N);
        k_scores<<<nodeBlocks, 256, 0, stream>>>(hlin, lasrc[L], ladst[L], sbuf, dbuf, N);
        k_gat_aggregate<<<nodeBlocks, 256, 0, stream>>>(rowptr, srcs, sbuf, dbuf,
                                                        hlin, lb[L], agg, N);
        k_zero<<<1, 256, 0, stream>>>(stats, 256);
        k_bn_stats<<<nb, 256, 0, stream>>>(agg, stats, N);
        k_bn_finalize<<<1, 128, 0, stream>>>(stats, gammas + (size_t)L * 128,
                                             betas + (size_t)L * 128, scsh, invN);
        if (L < 5)
            k_bn_apply<true ><<<cdiv(N * 32, 256), 256, 0, stream>>>(agg, scsh, resb, hbf, N);
        else
            k_bn_apply<false><<<cdiv(N * 32, 256), 256, 0, stream>>>(agg, scsh, nullptr, hbf, N);
    }

    // out = h @ Wo + bo
    k_gemm_bf16<<<N / 16, 256, 0, stream>>>(hbf, WoT, bo, (float*)d_out, N);
}
